// TDShaper_46170898432119
// MI455X (gfx1250) — matrix operand order repaired and checked by reading, not compiled, not run on hardware
//
#include <hip/hip_runtime.h>
#include <hip/hip_bf16.h>

typedef __attribute__((ext_vector_type(16))) _Float16 v16h;
typedef __attribute__((ext_vector_type(8)))  float    v8f;

#define B_BATCH 64
#define T_FRAMES 2000
#define FRAME 160
#define FEAT 128
#define ENVP 48                    // tenv dim 41 padded to 48 halves
#define BT (B_BATCH * T_FRAMES)    // 128000 rows, divisible by 16; T divisible by 16
#define K1STEPS 11                 // K = 256 (features t-1,t) + 96 (tenv padded) = 352
#define K2STEPS 10                 // K = 320 (alpha1 t-1,t)
#define NTILES 10                  // 160 / 16
#define LOG_EPS 1.52587890625e-05f

union AV { v16h v; _Float16 h[16]; unsigned u[8]; };

__device__ __forceinline__ unsigned short h_bits(float f) {
    union { _Float16 h; unsigned short u; } cv; cv.h = (_Float16)f; return cv.u;
}

// ---------------------------------------------------------------------------
// Envelope: one wave32 per frame. lane -> pooled idx {lane, 32+lane(<8)}.
// tenvp[frame][0..39] = e - avg, [40] = avg, [41..47] = 0 (K padding).
// ---------------------------------------------------------------------------
__global__ void env_kernel(const float* __restrict__ x, _Float16* __restrict__ tenvp) {
    int wid  = (blockIdx.x * blockDim.x + threadIdx.x) >> 5;   // frame id, grid exact
    int lane = threadIdx.x & 31;
    const float* xf = x + (size_t)wid * FRAME;

    float4 v0 = *(const float4*)(xf + lane * 4);
    float  m1 = (fabsf(v0.x) + fabsf(v0.y) + fabsf(v0.z) + fabsf(v0.w)) * 0.25f;
    float  e1 = logf(m1 + LOG_EPS);
    float  e2 = 0.f;
    if (lane < 8) {
        float4 v1 = *(const float4*)(xf + 128 + lane * 4);
        float  m2 = (fabsf(v1.x) + fabsf(v1.y) + fabsf(v1.z) + fabsf(v1.w)) * 0.25f;
        e2 = logf(m2 + LOG_EPS);
    }
    float s = e1 + e2;
    for (int o = 16; o > 0; o >>= 1) s += __shfl_xor(s, o, 32);
    float avg = s * (1.0f / 40.0f);

    _Float16* row = tenvp + (size_t)wid * ENVP;
    row[lane] = (_Float16)(e1 - avg);
    if (lane < 8)        row[32 + lane] = (_Float16)(e2 - avg);
    else if (lane == 8)  row[40]        = (_Float16)avg;
    else if (lane < 16)  row[32 + lane] = (_Float16)0.f;   // 41..47 zero pad
}

// ---------------------------------------------------------------------------
// Pack weights into CDNA5 16-bit B-fragment layout.
// B (32x16 K-slab): lane = (N & 15), K-half = lane>>4; VGPR j holds K=2j+p+8*half+(j>=4?8:0).
// Stage1 B[k][n]: k<128 -> W1f[n][k][0]; k<256 -> W1f[n][k-128][1];
//                 k>=256: kp=k-256, side=kp>=48, c=kp%48, W1t[n][c][side] (c<41 else 0)
// Stage2 B[k][n]: side=k>=160, c=k-(side?160:0), W2[n][c][side]
// 210 fragments total, one wave each.
// ---------------------------------------------------------------------------
__global__ void prep_kernel(const float* __restrict__ W1f, const float* __restrict__ W1t,
                            const float* __restrict__ W2,
                            unsigned* __restrict__ wpack1, unsigned* __restrict__ wpack2) {
    int wid  = (blockIdx.x * blockDim.x + threadIdx.x) >> 5;
    int lane = threadIdx.x & 31;
    if (wid >= NTILES * K1STEPS + NTILES * K2STEPS) return;   // wave-uniform exit

    int stage1 = (wid < NTILES * K1STEPS);
    int nT, ks;
    if (stage1) { nT = wid / K1STEPS; ks = wid % K1STEPS; }
    else        { int w = wid - NTILES * K1STEPS; nT = w / K2STEPS; ks = w % K2STEPS; }

    int n    = nT * 16 + (lane & 15);
    int half = lane >> 4;

    unsigned dw[8];
#pragma unroll
    for (int j = 0; j < 8; ++j) {
        float f[2];
#pragma unroll
        for (int p = 0; p < 2; ++p) {
            int   k = ks * 32 + 2 * j + p + 8 * half + (j >= 4 ? 8 : 0);
            float w = 0.f;
            if (stage1) {
                if (k < 128)      w = W1f[(n * 128 + k) * 2 + 0];
                else if (k < 256) w = W1f[(n * 128 + (k - 128)) * 2 + 1];
                else {
                    int kp = k - 256; int side = (kp >= 48); int c = kp - (side ? 48 : 0);
                    if (c < 41) w = W1t[(n * 41 + c) * 2 + side];
                }
            } else {
                int side = (k >= 160); int c = k - (side ? 160 : 0);
                w = W2[(n * 160 + c) * 2 + side];
            }
            f[p] = w;
        }
        dw[j] = (unsigned)h_bits(f[0]) | ((unsigned)h_bits(f[1]) << 16);
    }
    unsigned* dst = stage1 ? (wpack1 + ((size_t)(nT * K1STEPS + ks) * 32 + lane) * 8)
                           : (wpack2 + ((size_t)(nT * K2STEPS + ks) * 32 + lane) * 8);
    *(uint4*)(dst)     = make_uint4(dw[0], dw[1], dw[2], dw[3]);
    *(uint4*)(dst + 4) = make_uint4(dw[4], dw[5], dw[6], dw[7]);
}

// ---------------------------------------------------------------------------
// Stage 1 GEMM: alpha1 = leakyrelu(A1 * B1 + bias). One wave per 16x32 tile.
// A 16-bit layout: lane m = lane&15 (row), VGPR j -> K = cj + 8*half + p,
// cj = (j<4 ? 2j : 2j+8). The +8*half term is folded into the base pointers;
// per unrolled step the prev/cur side is a compile-time constant, so all loads
// are unconditional; causal t=0 masking is done with cndmask/mul (EXEC stays 1s).
// ---------------------------------------------------------------------------
__global__ void gemm1_kernel(const float* __restrict__ features,
                             const _Float16* __restrict__ tenvp,
                             const unsigned* __restrict__ wpack1,
                             const float* __restrict__ b1f, const float* __restrict__ b1t,
                             _Float16* __restrict__ alpha1) {
    int wid  = (blockIdx.x * blockDim.x + threadIdx.x) >> 5;
    int lane = threadIdx.x & 31;
    int rt = wid / 5, cg = wid % 5;                 // row tile, col group (2 N-tiles)
    int r0 = rt * 16;
    int b  = r0 / T_FRAMES;
    int t0 = r0 % T_FRAMES;
    int m = lane & 15, half = lane >> 4;
    int t = t0 + m;
    int rowCur  = b * T_FRAMES + t;
    bool havePrev = (t > 0);
    int rowPrev = havePrev ? rowCur - 1 : rowCur;           // safe dummy when t==0
    float    pm  = havePrev ? 1.0f : 0.0f;
    unsigned msk = havePrev ? 0xFFFFFFFFu : 0u;

    const float2*   fP  = (const float2*)(features + (size_t)rowPrev * FEAT) + 4 * half;
    const float2*   fC  = (const float2*)(features + (size_t)rowCur  * FEAT) + 4 * half;
    const unsigned* tvP = (const unsigned*)(tenvp + (size_t)rowPrev * ENVP) + 4 * half;
    const unsigned* tvC = (const unsigned*)(tenvp + (size_t)rowCur  * ENVP) + 4 * half;

    v8f acc0 = {}; v8f acc1 = {};
    const unsigned* bp0 = wpack1 + ((size_t)((cg * 2 + 0) * K1STEPS) * 32 + lane) * 8;
    const unsigned* bp1 = wpack1 + ((size_t)((cg * 2 + 1) * K1STEPS) * 32 + lane) * 8;

    // features: K-steps 0..3 (t-1, masked) and 4..7 (t)
#pragma unroll
    for (int ks = 0; ks < 8; ++ks) {
        const float2* src = ((ks < 4) ? fP : fC) + (ks & 3) * 16;
        AV a;
#pragma unroll
        for (int j = 0; j < 8; ++j) {
            int d = (j < 4) ? j : j + 4;            // cj/2
            float2 v = src[d];
            float sx = v.x, sy = v.y;
            if (ks < 4) { sx *= pm; sy *= pm; }
            a.h[2 * j]     = (_Float16)sx;
            a.h[2 * j + 1] = (_Float16)sy;
        }
        v16h bf0 = *(const v16h*)(bp0 + ks * 256);
        v16h bf1 = *(const v16h*)(bp1 + ks * 256);
        acc0 = __builtin_amdgcn_wmma_f32_16x16x32_f16(false, a.v, false, bf0, (short)0, acc0, false, false);
        acc1 = __builtin_amdgcn_wmma_f32_16x16x32_f16(false, a.v, false, bf1, (short)0, acc1, false, false);
    }
    // tenv step 8: k' = 0..31 -> all prev side (masked)
    {
        AV a;
#pragma unroll
        for (int j = 0; j < 8; ++j) { int d = (j < 4) ? j : j + 4; a.u[j] = tvP[d] & msk; }
        v16h bf0 = *(const v16h*)(bp0 + 8 * 256);
        v16h bf1 = *(const v16h*)(bp1 + 8 * 256);
        acc0 = __builtin_amdgcn_wmma_f32_16x16x32_f16(false, a.v, false, bf0, (short)0, acc0, false, false);
        acc1 = __builtin_amdgcn_wmma_f32_16x16x32_f16(false, a.v, false, bf1, (short)0, acc1, false, false);
    }
    // tenv step 9: j<4 -> prev k'=32..47 (masked); j>=4 -> cur c=0..15
    {
        AV a;
#pragma unroll
        for (int j = 0; j < 4; ++j) a.u[j] = tvP[16 + j] & msk;
#pragma unroll
        for (int j = 4; j < 8; ++j) a.u[j] = tvC[j - 4];
        v16h bf0 = *(const v16h*)(bp0 + 9 * 256);
        v16h bf1 = *(const v16h*)(bp1 + 9 * 256);
        acc0 = __builtin_amdgcn_wmma_f32_16x16x32_f16(false, a.v, false, bf0, (short)0, acc0, false, false);
        acc1 = __builtin_amdgcn_wmma_f32_16x16x32_f16(false, a.v, false, bf1, (short)0, acc1, false, false);
    }
    // tenv step 10: all cur side, c = 16..47
    {
        AV a;
#pragma unroll
        for (int j = 0; j < 8; ++j) { int d = (j < 4) ? j : j + 4; a.u[j] = tvC[8 + d]; }
        v16h bf0 = *(const v16h*)(bp0 + 10 * 256);
        v16h bf1 = *(const v16h*)(bp1 + 10 * 256);
        acc0 = __builtin_amdgcn_wmma_f32_16x16x32_f16(false, a.v, false, bf0, (short)0, acc0, false, false);
        acc1 = __builtin_amdgcn_wmma_f32_16x16x32_f16(false, a.v, false, bf1, (short)0, acc1, false, false);
    }

    // epilogue: D layout -> row M = r + 8*half, col N = lane&15
    int n0 = cg * 32 + (lane & 15);
    int n1 = n0 + 16;
    float bias0  = b1f[n0] + b1t[n0];
    float bias1v = b1f[n1] + b1t[n1];
#pragma unroll
    for (int r = 0; r < 8; ++r) {
        int tr = t0 + r + 8 * half;
        size_t base = (size_t)(b * T_FRAMES + tr) * FRAME;
        float v0 = acc0[r] + bias0;  v0 = v0 > 0.f ? v0 : 0.2f * v0;
        float v1 = acc1[r] + bias1v; v1 = v1 > 0.f ? v1 : 0.2f * v1;
        alpha1[base + n0] = (_Float16)v0;
        alpha1[base + n1] = (_Float16)v1;
    }
}

// ---------------------------------------------------------------------------
// Stage 2 GEMM: y = exp(A2 * B2 + b2) * x. K = 320: steps 0..4 prev (masked),
// steps 5..9 cur. Boundary 160 = 5*32 never straddles a step.
// ---------------------------------------------------------------------------
__global__ void gemm2_kernel(const _Float16* __restrict__ alpha1,
                             const unsigned* __restrict__ wpack2,
                             const float* __restrict__ b2,
                             const float* __restrict__ x,
                             float* __restrict__ out) {
    int wid  = (blockIdx.x * blockDim.x + threadIdx.x) >> 5;
    int lane = threadIdx.x & 31;
    int rt = wid / 5, cg = wid % 5;
    int r0 = rt * 16;
    int b  = r0 / T_FRAMES;
    int t0 = r0 % T_FRAMES;
    int m = lane & 15, half = lane >> 4;
    int t = t0 + m;
    int rowCur  = b * T_FRAMES + t;
    bool havePrev = (t > 0);
    int rowPrev = havePrev ? rowCur - 1 : rowCur;
    unsigned msk = havePrev ? 0xFFFFFFFFu : 0u;

    const unsigned* aP = (const unsigned*)(alpha1 + (size_t)rowPrev * FRAME) + 4 * half;
    const unsigned* aC = (const unsigned*)(alpha1 + (size_t)rowCur  * FRAME) + 4 * half;

    v8f acc0 = {}; v8f acc1 = {};
    const unsigned* bp0 = wpack2 + ((size_t)((cg * 2 + 0) * K2STEPS) * 32 + lane) * 8;
    const unsigned* bp1 = wpack2 + ((size_t)((cg * 2 + 1) * K2STEPS) * 32 + lane) * 8;

#pragma unroll
    for (int ks = 0; ks < K2STEPS; ++ks) {
        AV a;
#pragma unroll
        for (int j = 0; j < 8; ++j) {
            int d = (j < 4) ? j : j + 4;
            a.u[j] = (ks < 5) ? (aP[ks * 16 + d] & msk)
                              :  aC[(ks - 5) * 16 + d];
        }
        v16h bf0 = *(const v16h*)(bp0 + ks * 256);
        v16h bf1 = *(const v16h*)(bp1 + ks * 256);
        acc0 = __builtin_amdgcn_wmma_f32_16x16x32_f16(false, a.v, false, bf0, (short)0, acc0, false, false);
        acc1 = __builtin_amdgcn_wmma_f32_16x16x32_f16(false, a.v, false, bf1, (short)0, acc1, false, false);
    }

    int n0 = cg * 32 + (lane & 15);
    int n1 = n0 + 16;
    float bias0  = b2[n0];
    float bias1v = b2[n1];
#pragma unroll
    for (int r = 0; r < 8; ++r) {
        int tr = t0 + r + 8 * half;
        size_t base = (size_t)(b * T_FRAMES + tr) * FRAME;
        float z0 = acc0[r] + bias0;
        float z1 = acc1[r] + bias1v;
        out[base + n0] = expf(z0) * x[base + n0];
        out[base + n1] = expf(z1) * x[base + n1];
    }
}

// ---------------------------------------------------------------------------
extern "C" void kernel_launch(void* const* d_in, const int* in_sizes, int n_in,
                              void* d_out, int out_size, void* d_ws, size_t ws_size,
                              hipStream_t stream) {
    const float* x    = (const float*)d_in[0];
    const float* feat = (const float*)d_in[1];
    const float* W1f  = (const float*)d_in[2];
    const float* b1f  = (const float*)d_in[3];
    const float* W1t  = (const float*)d_in[4];
    const float* b1t  = (const float*)d_in[5];
    const float* W2   = (const float*)d_in[6];
    const float* b2   = (const float*)d_in[7];
    float* out = (float*)d_out;

    char* ws = (char*)d_ws;
    _Float16* tenvp  = (_Float16*)ws;                                   // 12,288,000 B
    _Float16* alpha1 = (_Float16*)(ws + (size_t)BT * ENVP * 2);         // 40,960,000 B
    unsigned* wpack1 = (unsigned*)(ws + (size_t)BT * ENVP * 2
                                      + (size_t)BT * FRAME * 2);        // 112,640 B
    unsigned* wpack2 = wpack1 + (size_t)NTILES * K1STEPS * 256;         // 102,400 B

    // 1) envelope: 128000 waves, 8 per block
    env_kernel<<<BT / 8, 256, 0, stream>>>(x, tenvp);
    // 2) weight packing: 210 waves
    prep_kernel<<<27, 256, 0, stream>>>(W1f, W1t, W2, wpack1, wpack2);
    // 3) + 4) GEMMs: (BT/16 row tiles) * 5 col groups = 40000 waves = 5000 blocks
    int blocks = (BT / 16) * 5 / 8;
    gemm1_kernel<<<blocks, 256, 0, stream>>>(feat, tenvp, wpack1, b1f, b1t, alpha1);
    gemm2_kernel<<<blocks, 256, 0, stream>>>(alpha1, wpack2, b2, x, out);
}
